// MPNNConvolution_47974784696370
// MI455X (gfx1250) — matrix operand order repaired and checked by reading, not compiled, not run on hardware
//
#include <hip/hip_runtime.h>
#include <hip/hip_bf16.h>
#include <math.h>

// ---------------------------------------------------------------------------
// Problem constants (from reference)
// ---------------------------------------------------------------------------
#define N_NODES  10000
#define N_EDGES  640000
#define HID      32
#define EDIM     128

// WMMA vector types (CDNA5, wave32)
typedef __attribute__((ext_vector_type(16))) __bf16 v16bf;
typedef __attribute__((ext_vector_type(8)))  float  v8f;
typedef __attribute__((ext_vector_type(4)))  float  f4v;   // for nontemporal b128 loads

// Fragment-buffer geometry: B-fragments for 16x16x32 bf16 WMMA.
//   frags 0..7   : w1      (4 K-chunks x 2 N-tiles)           h1 = relu(EA@w1+b1)
//   frags 8..71  : W2P     (32 K-chunks x 2 N-tiles)          msg = outer@W2P
//   frags 72..73 : B2T     (1 K-chunk  x 2 N-tiles)           + neigh@B2T (b2 term)
// Each fragment: 32 lanes x 16 bf16 (lane L holds N = L&15, K = 8*(L>>4)+e, plus 8 when e>=8).
#define N_FRAGS      74
#define FRAG_ELEMS   (N_FRAGS * 512)          // 37888 bf16
#define FRAG_BYTES   (FRAG_ELEMS * 2)         // 75776 B

// Workspace layout (bytes)
#define WS_AGG_OFF   0                        // N_NODES*HID f32      = 1,280,000
#define WS_FRAG_OFF  1280000                  // FRAG_BYTES           =    75,776
#define WS_MX_OFF    (1280000 + FRAG_BYTES)   // N_NODES*96 f32       = 3,840,000

// Dynamic LDS for the fused edge kernel
#define H1_BYTES     (8 * 512 * 4)            // 8 waves x [16 m][32 w] f32 (transposed)
#define ROWS_BYTES   (8 * 16 * 4)
#define SMEM_BYTES   (FRAG_BYTES + H1_BYTES + ROWS_BYTES)   // 92,672 B (< 320KB/WGP)

// ---------------------------------------------------------------------------
// Zero-init kernel (agg must start at 0 every call; harness poisons ws)
// ---------------------------------------------------------------------------
__global__ void mpnn_zero(float* __restrict__ p, int n) {
    int i = blockIdx.x * 256 + threadIdx.x;
    if (i < n) p[i] = 0.0f;
}

// ---------------------------------------------------------------------------
// Pack w1 / w2 / b2 into bf16 B-fragments in exact WMMA lane order.
//   W2P[w*32+j, i] = w2[w, i*32+j] ; B2T[j, i] = b2[i*32+j]
// ---------------------------------------------------------------------------
__global__ void mpnn_prep_frags(const float* __restrict__ w1,
                                const float* __restrict__ w2,
                                const float* __restrict__ b2,
                                __bf16* __restrict__ frag) {
    int idx = blockIdx.x * 256 + threadIdx.x;
    if (idx >= FRAG_ELEMS) return;
    int fragId = idx >> 9;          // /512
    int rem    = idx & 511;
    int lane   = rem >> 4;
    int e      = rem & 15;
    int n      = lane & 15;                       // N within 16-tile
    int k      = (e < 8) ? ((lane < 16) ? 0 : 8) + e : 16 + ((lane < 16) ? 0 : 8) + (e - 8);      // K within 32-chunk (B layout)
    float v;
    if (fragId < 8) {                             // w1 [128][32]
        int c = fragId >> 1, t = fragId & 1;
        v = w1[(c * 32 + k) * 32 + (t * 16 + n)];
    } else if (fragId < 72) {                     // w2 [32][1024], permuted
        int w = (fragId - 8) >> 1, t = (fragId - 8) & 1;
        v = w2[w * 1024 + (t * 16 + n) * 32 + k];
    } else {                                      // b2 [1024], permuted
        int t = fragId - 72;
        v = b2[(t * 16 + n) * 32 + k];
    }
    frag[idx] = (__bf16)v;
}

// ---------------------------------------------------------------------------
// Fused edge kernel: per wave, 16 edges.
//   Stage A: h1[16,32] = relu(edge_attr[16,128] @ w1 + b1)      (8 WMMA)
//   Stage B: msg[16,32] = Σ_w (h1[:,w] ⊙ neigh) @ W2P_w         (64 WMMA)
//            + neigh @ B2T                                      (2 WMMA)
//   Scatter: atomicAdd into agg[row].
// Stage-B is software-pipelined: next w's B-fragments are ds-loaded before the
// current WMMAs issue, hiding LDS latency behind the matrix pipe.
// ---------------------------------------------------------------------------
__global__ __launch_bounds__(256)
void mpnn_edge_wmma(const float* __restrict__ x,
                    const int*   __restrict__ eidx,
                    const float* __restrict__ ea,
                    const float* __restrict__ b1,
                    const __bf16* __restrict__ fragG,
                    float* __restrict__ agg) {
    extern __shared__ __align__(16) char smem[];
    __bf16* sfrag   = (__bf16*)smem;
    float*  h1all   = (float*)(smem + FRAG_BYTES);
    int*    rowsAll = (int*)(smem + FRAG_BYTES + H1_BYTES);

    // Cooperative copy of all B fragments into LDS (75776 B / 16 = 4736 uint4)
    {
        const uint4* src = (const uint4*)fragG;
        uint4*       dst = (uint4*)smem;
        for (int i = threadIdx.x; i < FRAG_BYTES / 16; i += 256) dst[i] = src[i];
    }

    const int wave = threadIdx.x >> 5;
    const int L    = threadIdx.x & 31;
    const int m    = L & 15;          // edge row within tile (also C-frag column)
    const int hi   = (L >> 4) & 1;    // lane half selects K subset of A-frags

    float* h1t  = h1all + wave * 512;     // [16 m][32 w]  (transposed for bulk reads)
    int*   rows = rowsAll + wave * 16;

    const int edge0 = (blockIdx.x * 8 + wave) * 16;   // 5000 blocks x 8 waves x 16 = E
    const int edge  = edge0 + m;
    const int colIdx = eidx[N_EDGES + edge];          // edge_index[1][e]
    if (L < 16) rows[m] = eidx[edge];                 // edge_index[0][e]

    // Gather neighbor features early (x is small & L2-resident; regular loads)
    const f4v* xv = (const f4v*)(x + (long long)colIdx * HID);
    f4v n0 = xv[hi ? 2 : 0], n1 = xv[(hi ? 2 : 0) + 1];
    f4v n2 = xv[hi ? 6 : 4], n3 = xv[(hi ? 6 : 4) + 1];
    float nv[16] = {n0.x,n0.y,n0.z,n0.w, n1.x,n1.y,n1.z,n1.w,
                    n2.x,n2.y,n2.z,n2.w, n3.x,n3.y,n3.z,n3.w};

    __syncthreads();   // B-fragments resident in LDS before any wave consumes them

    // ---- Stage A: h1 = relu(EA @ w1 + b1), K = 128 in 4 chunks of 32 ----
    // edge_attr is streamed exactly once -> nontemporal loads (protect L2 for x/agg)
    v8f c0 = {}, c1 = {};
    const f4v* ea4 = (const f4v*)(ea + (long long)edge * EDIM);
    #pragma unroll
    for (int c = 0; c < 4; ++c) {
        // A layout (16-bit, 16x32): lane half hi=0 -> K {0..7,16..23}; hi=1 -> {8..15,24..31}
        f4v r0 = __builtin_nontemporal_load(ea4 + c * 8 + (hi ? 2 : 0));
        f4v r1 = __builtin_nontemporal_load(ea4 + c * 8 + (hi ? 2 : 0) + 1);
        f4v r2 = __builtin_nontemporal_load(ea4 + c * 8 + (hi ? 6 : 4));
        f4v r3 = __builtin_nontemporal_load(ea4 + c * 8 + (hi ? 6 : 4) + 1);
        v16bf a;
        a[0]=(__bf16)r0.x; a[1]=(__bf16)r0.y; a[2]=(__bf16)r0.z; a[3]=(__bf16)r0.w;
        a[4]=(__bf16)r1.x; a[5]=(__bf16)r1.y; a[6]=(__bf16)r1.z; a[7]=(__bf16)r1.w;
        a[8]=(__bf16)r2.x; a[9]=(__bf16)r2.y; a[10]=(__bf16)r2.z; a[11]=(__bf16)r2.w;
        a[12]=(__bf16)r3.x; a[13]=(__bf16)r3.y; a[14]=(__bf16)r3.z; a[15]=(__bf16)r3.w;
        v16bf b0 = *(const v16bf*)(sfrag + (c * 2 + 0) * 512 + L * 16);
        v16bf b1f = *(const v16bf*)(sfrag + (c * 2 + 1) * 512 + L * 16);
        c0 = __builtin_amdgcn_wmma_f32_16x16x32_bf16(false, a, false, b0, (short)0, c0, false, false);
        c1 = __builtin_amdgcn_wmma_f32_16x16x32_bf16(false, a, false, b1f, (short)0, c1, false, false);
    }
    const float bia0 = b1[m], bia1 = b1[16 + m];
    #pragma unroll
    for (int r = 0; r < 8; ++r) {
        float v0 = c0[r] + bia0; c0[r] = v0 > 0.f ? v0 : 0.f;
        float v1 = c1[r] + bia1; c1[r] = v1 > 0.f ? v1 : 0.f;
    }
    // Stash h1 to LDS transposed [m][w]  (C layout: VGPR r -> M = r + 8*hi, N = L&15)
    #pragma unroll
    for (int r = 0; r < 8; ++r) {
        int mrow = r + (hi ? 8 : 0);
        h1t[mrow * 32 + m]      = c0[r];
        h1t[mrow * 32 + 16 + m] = c1[r];
    }

    __syncthreads();   // h1 + rows visible across lanes

    // Bulk-load this lane's 32 h1 scalars (row m) into VGPRs: 8x ds_load_b128
    float hs[32];
    #pragma unroll
    for (int q = 0; q < 8; ++q) {
        f4v t = *(const f4v*)(h1t + m * 32 + q * 4);
        hs[q * 4 + 0] = t.x; hs[q * 4 + 1] = t.y;
        hs[q * 4 + 2] = t.z; hs[q * 4 + 3] = t.w;
    }

    // ---- Stage B: messages, K = 1024 in 32 chunks; A_w = h1[:,w] ⊙ neigh ----
    // Software pipeline: prefetch next B-fragment pair while current WMMAs run.
    const __bf16* fb = sfrag + 8 * 512 + L * 16;      // W2P frags, this lane
    const __bf16* fbias = sfrag + 72 * 512 + L * 16;  // B2T frags, this lane
    v16bf b0  = *(const v16bf*)(fb);
    v16bf b1f = *(const v16bf*)(fb + 512);
    v8f m0 = {}, m1 = {};
    #pragma unroll
    for (int w = 0; w < 32; ++w) {
        const __bf16* nxt = (w < 31) ? (fb + (w + 1) * 1024) : fbias;
        v16bf nb0 = *(const v16bf*)(nxt);
        v16bf nb1 = *(const v16bf*)(nxt + 512);
        float s = hs[w];
        v16bf a;
        #pragma unroll
        for (int e = 0; e < 16; ++e) a[e] = (__bf16)(nv[e] * s);
        m0 = __builtin_amdgcn_wmma_f32_16x16x32_bf16(false, a, false, b0,  (short)0, m0, false, false);
        m1 = __builtin_amdgcn_wmma_f32_16x16x32_bf16(false, a, false, b1f, (short)0, m1, false, false);
        b0 = nb0; b1f = nb1;
    }
    {   // b2 bias term: + neigh @ B2T (fragments already in b0/b1f)
        v16bf a;
        #pragma unroll
        for (int e = 0; e < 16; ++e) a[e] = (__bf16)nv[e];
        m0 = __builtin_amdgcn_wmma_f32_16x16x32_bf16(false, a, false, b0,  (short)0, m0, false, false);
        m1 = __builtin_amdgcn_wmma_f32_16x16x32_bf16(false, a, false, b1f, (short)0, m1, false, false);
    }

    // ---- Scatter-add (tf.scatter_nd semantics: duplicates sum) ----
    #pragma unroll
    for (int r = 0; r < 8; ++r) {
        int mrow = r + (hi ? 8 : 0);
        int rw   = rows[mrow];
        atomicAdd(agg + (long long)rw * HID + m,      m0[r]);
        atomicAdd(agg + (long long)rw * HID + 16 + m, m1[r]);
    }
}

// ---------------------------------------------------------------------------
// MX = agg @ gru_kernel + b_in  (h-independent half of the GRU, parallel)
// ---------------------------------------------------------------------------
__global__ void mpnn_mx(const float* __restrict__ agg,
                        const float* __restrict__ gk,
                        const float* __restrict__ gbias,
                        float* __restrict__ MX) {
    int idx = blockIdx.x * 256 + threadIdx.x;
    if (idx >= N_NODES * 96) return;
    int n = idx / 96, o = idx % 96;
    float acc = gbias[o];                     // gru_bias[0][o]
    const float* arow = agg + n * HID;
    #pragma unroll
    for (int k = 0; k < HID; ++k) acc += arow[k] * gk[k * 96 + o];
    MX[idx] = acc;
}

// ---------------------------------------------------------------------------
// Serial GRU scan (reset_after=True), single wave32: lane o owns h[o].
// Recurrent kernel held in 96 VGPRs/lane; h replicated via LDS each step.
// Single-wave workgroup: barrier ops degrade to S_NOP (ISA 00_overview §3.1),
// so per-step sync cost is just the LDS waitcnt.
// ---------------------------------------------------------------------------
__global__ __launch_bounds__(32)
void mpnn_gru(const float* __restrict__ MX,
              const float* __restrict__ rk,
              const float* __restrict__ gbias,
              const float* __restrict__ h0,
              float* __restrict__ out) {
    __shared__ float hbuf[32];
    const int o = threadIdx.x;
    float a0[32], a1[32], a2[32];
    #pragma unroll
    for (int k = 0; k < 32; ++k) {
        a0[k] = rk[k * 96 + o];
        a1[k] = rk[k * 96 + 32 + o];
        a2[k] = rk[k * 96 + 64 + o];
    }
    const float br0 = gbias[96 + o], br1 = gbias[96 + 32 + o], br2 = gbias[96 + 64 + o];
    float hself = h0[o];
    hbuf[o] = hself;
    __syncthreads();
    float hv[32];
    #pragma unroll
    for (int k = 0; k < 32; ++k) hv[k] = hbuf[k];
    __syncthreads();

    for (int n = 0; n < N_NODES; ++n) {
        float mrz = br0, mrr = br1, mrh = br2;
        #pragma unroll
        for (int k = 0; k < 32; ++k) {
            float hk = hv[k];
            mrz += hk * a0[k];
            mrr += hk * a1[k];
            mrh += hk * a2[k];
        }
        const float* mx = MX + n * 96;
        float z  = 1.0f / (1.0f + expf(-(mx[o] + mrz)));
        float r  = 1.0f / (1.0f + expf(-(mx[32 + o] + mrr)));
        float hh = tanhf(mx[64 + o] + r * mrh);       // v_tanh_f32 (TRANS)
        float hn = z * hself + (1.0f - z) * hh;
        out[n * HID + o] = hn;
        hself = hn;
        hbuf[o] = hn;
        __syncthreads();
        #pragma unroll
        for (int k = 0; k < 32; ++k) hv[k] = hbuf[k];
        __syncthreads();
    }
    out[N_NODES * HID + o] = hself;                   // h_final
}

// ---------------------------------------------------------------------------
// Host launcher
// ---------------------------------------------------------------------------
extern "C" void kernel_launch(void* const* d_in, const int* in_sizes, int n_in,
                              void* d_out, int out_size, void* d_ws, size_t ws_size,
                              hipStream_t stream) {
    const float* x   = (const float*)d_in[0];
    const int*   eix = (const int*)  d_in[1];
    const float* ea  = (const float*)d_in[2];
    const float* h0  = (const float*)d_in[3];
    const float* w1  = (const float*)d_in[4];
    const float* b1  = (const float*)d_in[5];
    const float* w2  = (const float*)d_in[6];
    const float* b2  = (const float*)d_in[7];
    const float* gk  = (const float*)d_in[8];
    const float* grk = (const float*)d_in[9];
    const float* gb  = (const float*)d_in[10];
    float* out = (float*)d_out;

    char*   ws   = (char*)d_ws;
    float*  agg  = (float*)(ws + WS_AGG_OFF);     // [N_NODES, 32]
    __bf16* frag = (__bf16*)(ws + WS_FRAG_OFF);   // packed B fragments
    float*  MX   = (float*)(ws + WS_MX_OFF);      // [N_NODES, 96]

    mpnn_zero<<<(N_NODES * HID + 255) / 256, 256, 0, stream>>>(agg, N_NODES * HID);
    mpnn_prep_frags<<<(FRAG_ELEMS + 255) / 256, 256, 0, stream>>>(w1, w2, b2, frag);

    // 92,672 B dynamic LDS (> default 64KB cap; gfx1250 WGP has 320KB)
    (void)hipFuncSetAttribute((const void*)mpnn_edge_wmma,
                              hipFuncAttributeMaxDynamicSharedMemorySize,
                              (int)SMEM_BYTES);
    mpnn_edge_wmma<<<N_EDGES / 128, 256, SMEM_BYTES, stream>>>(x, eix, ea, b1, frag, agg);

    mpnn_mx<<<(N_NODES * 96 + 255) / 256, 256, 0, stream>>>(agg, gk, gb, MX);
    mpnn_gru<<<1, 32, 0, stream>>>(MX, grk, gb, h0, out);
}
